// Model_71322226917751
// MI455X (gfx1250) — matrix operand order repaired and checked by reading, not compiled, not run on hardware
//
#include <hip/hip_runtime.h>
#include <hip/hip_bf16.h>
#include <math.h>

// ---------------------------------------------------------------------------
// Problem constants (match reference)
// ---------------------------------------------------------------------------
#define CB   16      // batch
#define CTE  128     // encoder T
#define CTD  64      // decoder T
#define CH   256     // hidden
#define CE   256     // embed
#define COUT 32000   // vocab out
#define CG3  768     // 3*H

// ---------------------------------------------------------------------------
// WMMA types / helpers (CDNA5 gfx1250, wave32)
// ---------------------------------------------------------------------------
typedef __attribute__((ext_vector_type(16))) __bf16 v16bf;
typedef __attribute__((ext_vector_type(8)))  float  v8f;

union ABFrag { uint4 q[2]; v16bf v; };

__device__ __forceinline__ unsigned short f2b(float f) {
    unsigned u = __float_as_uint(f);
    unsigned r = u + 0x7FFFu + ((u >> 16) & 1u);   // round-to-nearest-even
    if (((u >> 23) & 0xFFu) == 0xFFu) r = u;       // inf/nan passthrough
    return (unsigned short)(r >> 16);
}

__device__ __forceinline__ v8f wmma_bf16(v16bf a, v16bf b, v8f c) {
    return __builtin_amdgcn_wmma_f32_16x16x32_bf16(
        /*neg_a=*/false, a, /*neg_b=*/false, b,
        /*c_mod=*/(short)0, c, /*reuse_a=*/false, /*reuse_b=*/false);
}

// A fragment: A is M(16) x K row-major; lane = {m = lane&15, khalf = lane>>4}
// lane holds K = khalf*8 + [0..7] in v[0..7] and khalf*8 + 16 + [0..7] in v[8..15]
__device__ __forceinline__ v16bf loadA_g(const unsigned short* __restrict__ A,
                                         int lda, int mbase, int k0, int lane) {
    int m  = lane & 15;
    int kb = (lane >> 4) * 8;
    const unsigned short* p = A + (size_t)(mbase + m) * lda + k0 + kb;
    ABFrag t;
    t.q[0] = *(const uint4*)(p);
    t.q[1] = *(const uint4*)(p + 16);
    return t.v;
}

// B fragment: computing X @ W^T with W row-major (N x K); B[k][n] = W[n][k].
// lane = {n = lane&15, khalf = lane>>4}; lane holds K = khalf*8 + [0..7] in v[0..7]
// and khalf*8 + 16 + [0..7] in v[8..15] of row n of W -> two 16B loads.
__device__ __forceinline__ v16bf loadB_w(const unsigned short* __restrict__ W,
                                         int ldw, int nbase, int k0, int lane) {
    int n  = lane & 15;
    int kh = (lane >> 4) * 8;
    const unsigned short* p = W + (size_t)(nbase + n) * ldw + k0 + kh;
    ABFrag t;
    t.q[0] = *(const uint4*)(p);
    t.q[1] = *(const uint4*)(p + 16);
    return t.v;
}

// ---------------------------------------------------------------------------
// f32 -> bf16 weight conversion
// ---------------------------------------------------------------------------
__global__ void f32_to_bf16_kernel(const float* __restrict__ in,
                                   unsigned short* __restrict__ out, int n) {
    int g = blockIdx.x * blockDim.x + threadIdx.x;
    if (g < n) out[g] = f2b(in[g]);
}

// ---------------------------------------------------------------------------
// Embedding gather -> bf16 activations, (nTok, 256)
// ---------------------------------------------------------------------------
__global__ void embed_bf16_kernel(const int* __restrict__ ids,
                                  const float* __restrict__ emb,
                                  unsigned short* __restrict__ out, int n) {
    int g = blockIdx.x * blockDim.x + threadIdx.x;
    if (g >= n) return;
    int tok = g >> 8;
    int e   = g & 255;
    out[g] = f2b(emb[(size_t)ids[tok] * CE + e]);
}

// ---------------------------------------------------------------------------
// Generic bf16 GEMM: C(MxN) = act( A(MxK) @ W(NxK)^T + bias )
// One wave computes a 64x16 tile (4 M-tiles reuse one B fragment).
// M % 64 == 0, N % 16 == 0, K % 32 == 0.
// ---------------------------------------------------------------------------
__global__ void __launch_bounds__(256)
gemm_bf16_kernel(const unsigned short* __restrict__ A, int lda,
                 const unsigned short* __restrict__ W, int ldw,
                 const float* __restrict__ bias,
                 float* __restrict__ Cf, int ldcf,
                 unsigned short* __restrict__ Cb, int ldcb,
                 int M, int N, int K, int act) {
    int lane = threadIdx.x & 31;
    int wid  = (blockIdx.x * blockDim.x + threadIdx.x) >> 5;
    int nT   = N >> 4;
    int total = (M >> 6) * nT;
    if (wid >= total) return;
    int mB = wid / nT;
    int nt = wid % nT;

    v8f acc0 = {}, acc1 = {}, acc2 = {}, acc3 = {};
    for (int k0 = 0; k0 < K; k0 += 32) {
        v16bf b  = loadB_w(W, ldw, nt * 16, k0, lane);
        v16bf a0 = loadA_g(A, lda, mB * 64 +  0, k0, lane);
        v16bf a1 = loadA_g(A, lda, mB * 64 + 16, k0, lane);
        v16bf a2 = loadA_g(A, lda, mB * 64 + 32, k0, lane);
        v16bf a3 = loadA_g(A, lda, mB * 64 + 48, k0, lane);
        acc0 = wmma_bf16(a0, b, acc0);
        acc1 = wmma_bf16(a1, b, acc1);
        acc2 = wmma_bf16(a2, b, acc2);
        acc3 = wmma_bf16(a3, b, acc3);
    }

    int n     = nt * 16 + (lane & 15);
    int mhalf = (lane >> 4) * 8;
    float bv  = bias ? bias[n] : 0.0f;
    v8f accs[4] = {acc0, acc1, acc2, acc3};
#pragma unroll
    for (int i = 0; i < 4; ++i) {
#pragma unroll
        for (int r = 0; r < 8; ++r) {
            int row = mB * 64 + i * 16 + mhalf + r;
            float v = accs[i][r] + bv;
            if (act == 1) v = tanhf(v);
            if (Cf) Cf[(size_t)row * ldcf + n] = v;
            if (Cb) Cb[(size_t)row * ldcb + n] = f2b(v);
        }
    }
}

// ---------------------------------------------------------------------------
// Persistent sequential GRU. One workgroup (16 waves, 512 threads).
// Wave w owns hidden columns [16w, 16w+16) of all three gates, so the gate
// nonlinearity and h update are fully in-register.
// GI: precomputed x@Wih^T + bih, shape (B*T, 768), row = b*T + t.
// ---------------------------------------------------------------------------
__global__ void __launch_bounds__(512)
gru_seq_kernel(const float* __restrict__ GI,
               const unsigned short* __restrict__ WhhB,  // (768,256) bf16
               const float* __restrict__ bhh,            // (768)
               const int* __restrict__ lengths,          // (B)
               const float* __restrict__ h_init,         // (B,H) or null
               float* __restrict__ outF,                 // (B,T,H) or null
               unsigned short* __restrict__ outB, int ldB,
               float* __restrict__ hT_out, int T) {
    __shared__ float          hF[CB * CH];
    __shared__ unsigned short hB[CB * CH];

    int tid  = threadIdx.x;
    int lane = tid & 31;
    int w    = tid >> 5;

    for (int i = tid; i < CB * CH; i += 512) {
        float v = h_init ? h_init[i] : 0.0f;
        hF[i] = v;
        hB[i] = f2b(v);
    }

    int col   = w * 16 + (lane & 15);
    int mhalf = (lane >> 4) * 8;
    float bh_r = bhh[col], bh_z = bhh[CH + col], bh_n = bhh[2 * CH + col];
    int len8[8];
#pragma unroll
    for (int r = 0; r < 8; ++r) len8[r] = lengths[mhalf + r];

    for (int t = 0; t < T; ++t) {
        __syncthreads();  // h writes of step t-1 (or init) visible
        // Load all A fragments (h as bf16) from LDS: shared across 3 gates.
        v16bf af[8];
        {
            int m  = lane & 15;
            int kb = (lane >> 4) * 8;
            const unsigned short* p = hB + m * CH + kb;
#pragma unroll
            for (int kk = 0; kk < 8; ++kk) {
                ABFrag tt;
                tt.q[0] = *(const uint4*)(p + kk * 32);
                tt.q[1] = *(const uint4*)(p + kk * 32 + 16);
                af[kk] = tt.v;
            }
        }
        __syncthreads();  // all waves done reading old h before anyone writes

        v8f acc[3];
#pragma unroll
        for (int g = 0; g < 3; ++g) {
            v8f c = {};
#pragma unroll
            for (int kk = 0; kk < 8; ++kk) {
                v16bf b = loadB_w(WhhB, CH, g * CH + w * 16, kk * 32, lane);
                c = wmma_bf16(af[kk], b, c);
            }
            acc[g] = c;
        }

#pragma unroll
        for (int r = 0; r < 8; ++r) {
            int m = mhalf + r;  // batch row
            size_t gib = ((size_t)m * T + t) * CG3 + col;
            float gr = GI[gib], gz = GI[gib + CH], gn = GI[gib + 2 * CH];
            float ghr = acc[0][r] + bh_r;
            float ghz = acc[1][r] + bh_z;
            float ghn = acc[2][r] + bh_n;
            float rg = 1.0f / (1.0f + __expf(-(gr + ghr)));
            float zg = 1.0f / (1.0f + __expf(-(gz + ghz)));
            float nn = tanhf(gn + rg * ghn);
            float hold = hF[m * CH + col];
            float hnew = (1.0f - zg) * nn + zg * hold;
            bool valid = (t < len8[r]);
            float hsel = valid ? hnew : hold;
            float ov   = valid ? hnew : 0.0f;
            hF[m * CH + col] = hsel;
            hB[m * CH + col] = f2b(hsel);
            size_t orow = (size_t)m * T + t;
            if (outF) outF[orow * CH + col] = ov;
            outB[orow * ldB + col] = f2b(ov);
        }
    }
    __syncthreads();
    for (int i = tid; i < CB * CH; i += 512) hT_out[i] = hF[i];
}

// ---------------------------------------------------------------------------
// Attention: one block per (b, td). Energies e[te] = v . tanh(ep+dp'), exact
// softmax restricted to valid te (identical to reference after aw/s renorm),
// f32 context accumulation; bf16 context written into concat buffer col 256+.
// ---------------------------------------------------------------------------
__global__ void __launch_bounds__(256)
attention_kernel(const float* __restrict__ ep,    // (B*TE, H)
                 const float* __restrict__ dpw,   // (B*TD, H), includes Wb
                 const float* __restrict__ vvec,  // (H)
                 const float* __restrict__ encF,  // (B,TE,H) f32
                 const int* __restrict__ enc_len,
                 const int* __restrict__ dec_len,
                 unsigned short* __restrict__ catB) {  // (B*TD, 512)
    int bt = blockIdx.x;            // b*TD + td
    int b  = bt / CTD;
    int td = bt % CTD;
    int tid = threadIdx.x, lane = tid & 31, w = tid >> 5;

    __shared__ float s_dp[CH], s_v[CH], s_e[CTE], s_a[CTE], s_red[64];
    __shared__ float s_m, s_s;

    bool tdvalid = td < dec_len[b];
    int  L       = enc_len[b];
    s_dp[tid] = dpw[(size_t)bt * CH + tid];
    s_v[tid]  = vvec[tid];
    __syncthreads();

    // energies: wave w handles te in [w*16, w*16+16)
    for (int i = 0; i < 16; ++i) {
        int te = w * 16 + i;
        float part = 0.0f;
        if (te < L) {
            const float* er = ep + ((size_t)b * CTE + te) * CH;
#pragma unroll
            for (int j = 0; j < 8; ++j) {
                int h = lane + 32 * j;
                part += s_v[h] * tanhf(er[h] + s_dp[h]);
            }
        }
        for (int off = 16; off; off >>= 1) part += __shfl_down(part, off);
        if (lane == 0) s_e[te] = (te < L) ? part : -3.0e38f;
    }
    __syncthreads();

    if (tid < 64) s_red[tid] = fmaxf(s_e[tid], s_e[tid + 64]);
    __syncthreads();
    if (tid < 32) {
        float v = fmaxf(s_red[tid], s_red[tid + 32]);
        for (int off = 16; off; off >>= 1) v = fmaxf(v, __shfl_down(v, off));
        if (tid == 0) s_m = v;
    }
    __syncthreads();
    if (tid < CTE) s_a[tid] = (tid < L) ? __expf(s_e[tid] - s_m) : 0.0f;
    __syncthreads();
    if (tid < 64) s_red[tid] = s_a[tid] + s_a[tid + 64];
    __syncthreads();
    if (tid < 32) {
        float v = s_red[tid] + s_red[tid + 32];
        for (int off = 16; off; off >>= 1) v += __shfl_down(v, off);
        if (tid == 0) s_s = v;
    }
    __syncthreads();
    float inv = 1.0f / s_s;  // L >= 1 always

    float acc = 0.0f;
    if (tdvalid) {
        const float* eb = encF + ((size_t)b * CTE) * CH + tid;
        for (int te = 0; te < L; ++te) acc += (s_a[te] * inv) * eb[(size_t)te * CH];
    }
    catB[(size_t)bt * 512 + 256 + tid] = f2b(acc);
}

// ---------------------------------------------------------------------------
// Host-side orchestration
// ---------------------------------------------------------------------------
extern "C" void kernel_launch(void* const* d_in, const int* in_sizes, int n_in,
                              void* d_out, int out_size, void* d_ws, size_t ws_size,
                              hipStream_t stream) {
    (void)in_sizes; (void)n_in; (void)out_size; (void)ws_size;

    const int*   enc_in  = (const int*)d_in[0];
    const int*   enc_len = (const int*)d_in[1];
    const int*   dec_in  = (const int*)d_in[2];
    const int*   dec_len = (const int*)d_in[3];
    const float* emb     = (const float*)d_in[4];
    const float* eWih    = (const float*)d_in[5];
    const float* eWhh    = (const float*)d_in[6];
    const float* ebih    = (const float*)d_in[7];
    const float* ebhh    = (const float*)d_in[8];
    const float* dWih    = (const float*)d_in[9];
    const float* dWhh    = (const float*)d_in[10];
    const float* dbih    = (const float*)d_in[11];
    const float* dbhh    = (const float*)d_in[12];
    const float* attnW   = (const float*)d_in[13];
    const float* attnWb  = (const float*)d_in[14];
    const float* attnV   = (const float*)d_in[15];
    const float* denseW  = (const float*)d_in[17];
    const float* denseb  = (const float*)d_in[18];
    const float* outW    = (const float*)d_in[19];
    const float* outb    = (const float*)d_in[20];
    float* logits = (float*)d_out;

    // --- workspace carve (all offsets 256B aligned) ---
    char* base = (char*)d_ws;
    size_t off = 0;
    auto carve = [&](size_t bytes) -> void* {
        void* p = base + off;
        off = (off + bytes + 255) & ~(size_t)255;
        return p;
    };
    unsigned short* XencB   = (unsigned short*)carve((size_t)CB * CTE * CE * 2);
    unsigned short* XdecB   = (unsigned short*)carve((size_t)CB * CTD * CE * 2);
    unsigned short* WeIhB   = (unsigned short*)carve((size_t)CG3 * CE * 2);
    unsigned short* WeHhB   = (unsigned short*)carve((size_t)CG3 * CH * 2);
    unsigned short* WdIhB   = (unsigned short*)carve((size_t)CG3 * CE * 2);
    unsigned short* WdHhB   = (unsigned short*)carve((size_t)CG3 * CH * 2);
    unsigned short* WattnB  = (unsigned short*)carve((size_t)CH * 2 * CH * 2);
    unsigned short* WdenseB = (unsigned short*)carve((size_t)CH * 2 * CH * 2);
    unsigned short* WoutB   = (unsigned short*)carve((size_t)COUT * CH * 2);
    float*          GIenc   = (float*)carve((size_t)CB * CTE * CG3 * 4);
    float*          GIdec   = (float*)carve((size_t)CB * CTD * CG3 * 4);
    float*          EncF    = (float*)carve((size_t)CB * CTE * CH * 4);
    unsigned short* EncB    = (unsigned short*)carve((size_t)CB * CTE * CH * 2);
    unsigned short* CatB    = (unsigned short*)carve((size_t)CB * CTD * 2 * CH * 2);
    float*          epB     = (float*)carve((size_t)CB * CTE * CH * 4);
    float*          dpB     = (float*)carve((size_t)CB * CTD * CH * 4);
    unsigned short* DenseB  = (unsigned short*)carve((size_t)CB * CTD * CH * 2);
    float*          hT      = (float*)carve((size_t)CB * CH * 4);

    auto cvt = [&](const float* src, unsigned short* dst, int n) {
        f32_to_bf16_kernel<<<(n + 255) / 256, 256, 0, stream>>>(src, dst, n);
    };
    auto gemm = [&](const unsigned short* A, int lda, const unsigned short* W, int ldw,
                    const float* bias, float* Cf, int ldcf, unsigned short* Cb, int ldcb,
                    int M, int N, int K, int act) {
        int waves  = (M / 64) * (N / 16);
        int blocks = (waves + 7) / 8;
        gemm_bf16_kernel<<<blocks, 256, 0, stream>>>(A, lda, W, ldw, bias, Cf, ldcf,
                                                     Cb, ldcb, M, N, K, act);
    };

    // --- 1. weight conversion (bf16, L2-resident) ---
    cvt(eWih,   WeIhB,   CG3 * CE);
    cvt(eWhh,   WeHhB,   CG3 * CH);
    cvt(dWih,   WdIhB,   CG3 * CE);
    cvt(dWhh,   WdHhB,   CG3 * CH);
    cvt(attnW,  WattnB,  CH * 2 * CH);
    cvt(denseW, WdenseB, CH * 2 * CH);
    cvt(outW,   WoutB,   COUT * CH);

    // --- 2. embedding gathers -> bf16 ---
    {
        int n1 = CB * CTE * CE, n2 = CB * CTD * CE;
        embed_bf16_kernel<<<(n1 + 255) / 256, 256, 0, stream>>>(enc_in, emb, XencB, n1);
        embed_bf16_kernel<<<(n2 + 255) / 256, 256, 0, stream>>>(dec_in, emb, XdecB, n2);
    }

    // --- 3. batched input-gate GEMMs: GI = X @ Wih^T + bih ---
    gemm(XencB, CE, WeIhB, CE, ebih, GIenc, CG3, nullptr, 0, CB * CTE, CG3, CE, 0);
    gemm(XdecB, CE, WdIhB, CE, dbih, GIdec, CG3, nullptr, 0, CB * CTD, CG3, CE, 0);

    // --- 4. sequential GRUs (persistent single-WGP workgroups) ---
    gru_seq_kernel<<<1, 512, 0, stream>>>(GIenc, WeHhB, ebhh, enc_len, nullptr,
                                          EncF, EncB, CH, hT, CTE);
    gru_seq_kernel<<<1, 512, 0, stream>>>(GIdec, WdHhB, dbhh, dec_len, hT,
                                          nullptr, CatB, 2 * CH, hT, CTD);

    // --- 5. attention projections: ep = enc@We^T, dp = dec@Wd^T + Wb ---
    gemm(EncB, CH, WattnB, 2 * CH, nullptr, epB, CH, nullptr, 0, CB * CTE, CH, CH, 0);
    gemm(CatB, 2 * CH, WattnB + CH, 2 * CH, attnWb, dpB, CH, nullptr, 0,
         CB * CTD, CH, CH, 0);

    // --- 6. attention energies / softmax / context -> CatB[:,256:512] ---
    attention_kernel<<<CB * CTD, 256, 0, stream>>>(epB, dpB, attnV, EncF,
                                                   enc_len, dec_len, CatB);

    // --- 7. dense = tanh([dec_out, context] @ dense_W^T + b) -> bf16 ---
    gemm(CatB, 2 * CH, WdenseB, 2 * CH, denseb, nullptr, 0, DenseB, CH,
         CB * CTD, CH, 2 * CH, 1);

    // --- 8. logits = dense @ out_W^T + out_b -> f32 d_out ---
    gemm(DenseB, CH, WoutB, CH, outb, logits, COUT, nullptr, 0,
         CB * CTD, COUT, CH, 0);
}
